// LoraLinear_64295660421972
// MI455X (gfx1250) — matrix operand order repaired and checked by reading, not compiled, not run on hardware
//
#include <hip/hip_runtime.h>
#include <hip/hip_bf16.h>

// ---------------- problem constants (from reference) ----------------
#define D_IN    4096
#define D_OUT   4096
#define NROWS   8192
#define T_TASKS 10
#define RANK    16
#define SCALING (1.0f / 16.0f)

// ---------------- GEMM tiling ----------------
#define BM 128
#define BN 128
#define BK 32
#define PITCH 40   // LDS row pitch in halves (80B, 16B-aligned, bank-skewed)

typedef __attribute__((ext_vector_type(16))) __bf16        v16bf;
typedef __attribute__((ext_vector_type(8)))  float         v8f;
typedef __attribute__((ext_vector_type(8)))  unsigned int  v8u;

__device__ __forceinline__ unsigned short bf16_rne(float f) {
    unsigned int u = __builtin_bit_cast(unsigned int, f);
    unsigned int r = u + 0x7FFFu + ((u >> 16) & 1u);   // round-to-nearest-even
    return (unsigned short)(r >> 16);
}
__device__ __forceinline__ float bf16_f32(unsigned short h) {
    unsigned int u = ((unsigned int)h) << 16;
    return __builtin_bit_cast(float, u);
}

// ---- gfx1250 async copy: global -> LDS, 16B per lane, ASYNCcnt-tracked ----
// LDS flat addresses carry the workgroup-relative LDS byte offset in the low
// 32 bits (ISA 10.2), so truncation is the VDST.
__device__ __forceinline__ unsigned int lds_off_u32(const void* p) {
    return (unsigned int)(unsigned long long)p;
}
__device__ __forceinline__ void async_b128(unsigned int lds_byte_off, const void* gptr) {
    asm volatile("global_load_async_to_lds_b128 %0, %1, off"
                 :: "v"(lds_byte_off), "v"(gptr) : "memory");
}
__device__ __forceinline__ void wait_async0() {
    asm volatile("s_wait_asynccnt 0x0" ::: "memory");
}

// ====================================================================
// Kernel 1: W_eff = W + (1/r) * sum_t B_t @ A_t  -> split bf16 hi/lo
// ====================================================================
__global__ void __launch_bounds__(256)
build_weff(const float* __restrict__ W, const float* __restrict__ lA,
           const float* __restrict__ lB, unsigned short* __restrict__ whi,
           unsigned short* __restrict__ wlo) {
    __shared__ float sB[T_TASKS * RANK];
    const int o = blockIdx.y;
    const int i = blockIdx.x * 256 + threadIdx.x;

    if (threadIdx.x < T_TASKS * RANK) {
        const int t = threadIdx.x / RANK;
        const int r = threadIdx.x % RANK;
        sB[threadIdx.x] = SCALING * lB[((size_t)t * D_OUT + o) * RANK + r];
    }
    __syncthreads();

    float acc = W[(size_t)o * D_IN + i];
#pragma unroll 8
    for (int tr = 0; tr < T_TASKS * RANK; ++tr) {
        acc = fmaf(sB[tr], lA[(size_t)tr * D_IN + i], acc);
    }

    const unsigned short hi = bf16_rne(acc);
    whi[(size_t)o * D_IN + i] = hi;
    wlo[(size_t)o * D_IN + i] = bf16_rne(acc - bf16_f32(hi));
}

// ====================================================================
// Kernel 2: split x (f32) into bf16 hi/lo planes (one cheap pass)
// ====================================================================
__global__ void __launch_bounds__(256)
split_x(const float* __restrict__ X, unsigned short* __restrict__ xhi,
        unsigned short* __restrict__ xlo) {
    const size_t i4 = (size_t)blockIdx.x * 256 + threadIdx.x;   // float4 index
    const float4 v = ((const float4*)X)[i4];
    const unsigned short h0 = bf16_rne(v.x), h1 = bf16_rne(v.y);
    const unsigned short h2 = bf16_rne(v.z), h3 = bf16_rne(v.w);
    ushort4 hh; hh.x = h0; hh.y = h1; hh.z = h2; hh.w = h3;
    ushort4 ll;
    ll.x = bf16_rne(v.x - bf16_f32(h0));
    ll.y = bf16_rne(v.y - bf16_f32(h1));
    ll.z = bf16_rne(v.z - bf16_f32(h2));
    ll.w = bf16_rne(v.w - bf16_f32(h3));
    ((ushort4*)xhi)[i4] = hh;
    ((ushort4*)xlo)[i4] = ll;
}

// ====================================================================
// Shared fragment + WMMA compute stage (reads one LDS buffer of 4 planes)
// ====================================================================
__device__ __forceinline__ void
compute_tiles(const unsigned short* xh, const unsigned short* xl,
              const unsigned short* wh, const unsigned short* wl,
              int lane, int wm, int wn, v8f acc[4][2]) {
    const int mrow  = lane & 15;
    const int abase = (lane < 16) ? 0 : 8;    // A: K 0-7/16-23 vs 8-15/24-31
    const int bbase = (lane < 16) ? 0 : 8;    // B: K 0-7/16-23 vs 8-15/24-31

    v16bf ah[4], al[4], bh[2], bl[2];
#pragma unroll
    for (int sm = 0; sm < 4; ++sm) {
        const int rr = (wm * 64 + sm * 16 + mrow) * PITCH;
        uint4 q0 = *(const uint4*)&xh[rr + abase];
        uint4 q1 = *(const uint4*)&xh[rr + abase + 16];
        v8u u; u[0]=q0.x; u[1]=q0.y; u[2]=q0.z; u[3]=q0.w;
               u[4]=q1.x; u[5]=q1.y; u[6]=q1.z; u[7]=q1.w;
        ah[sm] = __builtin_bit_cast(v16bf, u);
        q0 = *(const uint4*)&xl[rr + abase];
        q1 = *(const uint4*)&xl[rr + abase + 16];
        u[0]=q0.x; u[1]=q0.y; u[2]=q0.z; u[3]=q0.w;
        u[4]=q1.x; u[5]=q1.y; u[6]=q1.z; u[7]=q1.w;
        al[sm] = __builtin_bit_cast(v16bf, u);
    }
#pragma unroll
    for (int sn = 0; sn < 2; ++sn) {
        const int cc = (wn * 32 + sn * 16 + (lane & 15)) * PITCH;
        uint4 q0 = *(const uint4*)&wh[cc + bbase];
        uint4 q1 = *(const uint4*)&wh[cc + bbase + 16];
        v8u u; u[0]=q0.x; u[1]=q0.y; u[2]=q0.z; u[3]=q0.w;
               u[4]=q1.x; u[5]=q1.y; u[6]=q1.z; u[7]=q1.w;
        bh[sn] = __builtin_bit_cast(v16bf, u);
        q0 = *(const uint4*)&wl[cc + bbase];
        q1 = *(const uint4*)&wl[cc + bbase + 16];
        u[0]=q0.x; u[1]=q0.y; u[2]=q0.z; u[3]=q0.w;
        u[4]=q1.x; u[5]=q1.y; u[6]=q1.z; u[7]=q1.w;
        bl[sn] = __builtin_bit_cast(v16bf, u);
    }

#pragma unroll
    for (int sm = 0; sm < 4; ++sm)
#pragma unroll
        for (int sn = 0; sn < 2; ++sn) {
            acc[sm][sn] = __builtin_amdgcn_wmma_f32_16x16x32_bf16(
                false, ah[sm], false, bh[sn], (short)0, acc[sm][sn], false, false);
            acc[sm][sn] = __builtin_amdgcn_wmma_f32_16x16x32_bf16(
                false, ah[sm], false, bl[sn], (short)0, acc[sm][sn], false, false);
            acc[sm][sn] = __builtin_amdgcn_wmma_f32_16x16x32_bf16(
                false, al[sm], false, bh[sn], (short)0, acc[sm][sn], false, false);
        }
}

__device__ __forceinline__ void
epilogue(v8f acc[4][2], const float* bias, float* out,
         int lane, int wm, int wn, int row0, int col0) {
    const int n    = lane & 15;
    const int mofs = (lane >> 4) << 3;   // 0 or 8
#pragma unroll
    for (int sn = 0; sn < 2; ++sn) {
        const int col = col0 + wn * 32 + sn * 16 + n;
        const float bv = bias[col];
#pragma unroll
        for (int sm = 0; sm < 4; ++sm) {
            const int mbase = row0 + wm * 64 + sm * 16 + mofs;
#pragma unroll
            for (int r = 0; r < 8; ++r) {
                out[(size_t)(mbase + r) * D_OUT + col] = acc[sm][sn][r] + bv;
            }
        }
    }
}

// ====================================================================
// Kernel 3a (fast path): all-bf16 GEMM, async-to-LDS double buffered
// ====================================================================
__global__ void __launch_bounds__(256)
lora_gemm_async(const unsigned short* __restrict__ xhi,
                const unsigned short* __restrict__ xlo,
                const unsigned short* __restrict__ whi,
                const unsigned short* __restrict__ wlo,
                const float* __restrict__ bias, float* __restrict__ out) {
    constexpr int PLANE = BM * PITCH;                 // halves per plane
    __shared__ __attribute__((aligned(16))) unsigned short lds[2 * 4 * PLANE];

    const int tid  = threadIdx.x;
    const int lane = tid & 31;
    const int wid  = tid >> 5;
    const int wm   = wid >> 2;
    const int wn   = wid & 3;
    const int row0 = blockIdx.x * BM;
    const int col0 = blockIdx.y * BN;

    // per-thread staging coordinates: 2 b128 slots per plane per thread
    const int r0 = tid >> 2;                          // rows for it=0 / it=1
    const int r1 = (256 + tid) >> 2;
    const int c0 = (tid & 3) << 3;                    // half-column 0,8,16,24

    auto stage = [&](int k0, int b) {
        const unsigned short* base = &lds[(b * 4) * PLANE];
#pragma unroll
        for (int it = 0; it < 2; ++it) {
            const int r    = it ? r1 : r0;
            const unsigned int loff = (unsigned int)(r * PITCH + c0);
            const size_t gx = (size_t)(row0 + r) * D_IN + k0 + c0;
            const size_t gw = (size_t)(col0 + r) * D_IN + k0 + c0;
            async_b128(lds_off_u32(base + 0 * PLANE + loff), xhi + gx);
            async_b128(lds_off_u32(base + 1 * PLANE + loff), xlo + gx);
            async_b128(lds_off_u32(base + 2 * PLANE + loff), whi + gw);
            async_b128(lds_off_u32(base + 3 * PLANE + loff), wlo + gw);
        }
    };

    v8f acc[4][2];
#pragma unroll
    for (int a = 0; a < 4; ++a)
#pragma unroll
        for (int c = 0; c < 2; ++c) acc[a][c] = v8f{};

    stage(0, 0);
    wait_async0();
    __syncthreads();

    const int NK = D_IN / BK;
    for (int k = 0; k < NK; ++k) {
        const int b = k & 1;
        if (k + 1 < NK) stage((k + 1) * BK, b ^ 1);

        const unsigned short* xh = &lds[(b * 4 + 0) * PLANE];
        const unsigned short* xl = &lds[(b * 4 + 1) * PLANE];
        const unsigned short* wh = &lds[(b * 4 + 2) * PLANE];
        const unsigned short* wl = &lds[(b * 4 + 3) * PLANE];
        compute_tiles(xh, xl, wh, wl, lane, wm, wn, acc);

        if (k + 1 < NK) {
            wait_async0();       // next tile landed in LDS
            __syncthreads();     // all waves done reading buffer b
        }
    }

    epilogue(acc, bias, out, lane, wm, wn, row0, col0);
}

// ====================================================================
// Kernel 3b (fallback, small ws): f32 x converted in-kernel (round-1 path)
// ====================================================================
__global__ void __launch_bounds__(256)
lora_gemm_fallback(const float* __restrict__ X, const unsigned short* __restrict__ whi,
                   const unsigned short* __restrict__ wlo, const float* __restrict__ bias,
                   float* __restrict__ out) {
    __shared__ __attribute__((aligned(16))) unsigned short lxh[BM * PITCH];
    __shared__ __attribute__((aligned(16))) unsigned short lxl[BM * PITCH];
    __shared__ __attribute__((aligned(16))) unsigned short lwh[BN * PITCH];
    __shared__ __attribute__((aligned(16))) unsigned short lwl[BN * PITCH];

    const int tid  = threadIdx.x;
    const int lane = tid & 31;
    const int wid  = tid >> 5;
    const int wm   = wid >> 2;
    const int wn   = wid & 3;
    const int row0 = blockIdx.x * BM;
    const int col0 = blockIdx.y * BN;

    v8f acc[4][2];
#pragma unroll
    for (int a = 0; a < 4; ++a)
#pragma unroll
        for (int c = 0; c < 2; ++c) acc[a][c] = v8f{};

    for (int k0 = 0; k0 < D_IN; k0 += BK) {
#pragma unroll
        for (int it = 0; it < 4; ++it) {
            const int idx = it * 256 + tid;
            const int r   = idx >> 3;
            const int c   = (idx & 7) << 2;
            const float4 v = *(const float4*)(X + (size_t)(row0 + r) * D_IN + k0 + c);
            const int off = r * PITCH + c;
            const unsigned short h0 = bf16_rne(v.x), h1 = bf16_rne(v.y);
            const unsigned short h2 = bf16_rne(v.z), h3 = bf16_rne(v.w);
            lxh[off + 0] = h0; lxh[off + 1] = h1; lxh[off + 2] = h2; lxh[off + 3] = h3;
            lxl[off + 0] = bf16_rne(v.x - bf16_f32(h0));
            lxl[off + 1] = bf16_rne(v.y - bf16_f32(h1));
            lxl[off + 2] = bf16_rne(v.z - bf16_f32(h2));
            lxl[off + 3] = bf16_rne(v.w - bf16_f32(h3));
        }
#pragma unroll
        for (int it = 0; it < 2; ++it) {
            const int idx = it * 256 + tid;
            const int r   = idx >> 2;
            const int c   = (idx & 3) << 3;
            const size_t g = (size_t)(col0 + r) * D_IN + k0 + c;
            *(uint4*)&lwh[r * PITCH + c] = *(const uint4*)(whi + g);
            *(uint4*)&lwl[r * PITCH + c] = *(const uint4*)(wlo + g);
        }
        __syncthreads();
        compute_tiles(lxh, lxl, lwh, lwl, lane, wm, wn, acc);
        __syncthreads();
    }

    epilogue(acc, bias, out, lane, wm, wn, row0, col0);
}

// ====================================================================
extern "C" void kernel_launch(void* const* d_in, const int* in_sizes, int n_in,
                              void* d_out, int out_size, void* d_ws, size_t ws_size,
                              hipStream_t stream) {
    const float* x  = (const float*)d_in[0];
    const float* W  = (const float*)d_in[1];
    const float* bv = (const float*)d_in[2];
    const float* lA = (const float*)d_in[3];
    const float* lB = (const float*)d_in[4];

    unsigned short* whi = (unsigned short*)d_ws;
    unsigned short* wlo = whi + (size_t)D_OUT * D_IN;
    unsigned short* xhi = wlo + (size_t)D_OUT * D_IN;
    unsigned short* xlo = xhi + (size_t)NROWS * D_IN;

    const size_t need_w = 2ull * D_OUT * D_IN * sizeof(unsigned short);
    const size_t need_x = 2ull * NROWS * D_IN * sizeof(unsigned short);

    dim3 g1(D_IN / 256, D_OUT);
    build_weff<<<g1, 256, 0, stream>>>(W, lA, lB, whi, wlo);

    dim3 g2(NROWS / BM, D_OUT / BN);
    if (ws_size >= need_w + need_x) {
        const size_t nf4 = (size_t)NROWS * D_IN / 4;
        split_x<<<dim3(nf4 / 256), 256, 0, stream>>>(x, xhi, xlo);
        lora_gemm_async<<<g2, 256, 0, stream>>>(xhi, xlo, whi, wlo, bv, (float*)d_out);
    } else {
        lora_gemm_fallback<<<g2, 256, 0, stream>>>(x, whi, wlo, bv, (float*)d_out);
    }
}
